// ImprovedFullContextTransformer_83829171683731
// MI455X (gfx1250) — matrix operand order repaired and checked by reading, not compiled, not run on hardware
//
#include <hip/hip_runtime.h>
#include <hip/hip_bf16.h>

// ---------------------------------------------------------------------------
// Problem constants (from reference): L=6, D=1024, H=16, DK=64, FF=4096,
// T=1024, B=2.  M = T*B = 2048 rows everywhere.
// ---------------------------------------------------------------------------
#define C_T   1024
#define C_B   2
#define C_D   1024
#define C_H   16
#define C_DK  64
#define C_FF  4096
#define C_L   6
#define C_M   (C_T * C_B)
#define C_SCALE 0.125f          /* 1/sqrt(64) */
#define C_NEGINF (-1e30f)

typedef __attribute__((ext_vector_type(16))) __bf16 v16bf;
typedef __attribute__((ext_vector_type(8)))  float  v8f;

// fragment container: 16 bf16 = 32 bytes = 2 x uint4
union ABFrag {
    uint4          u4[2];
    unsigned short h[16];
    v16bf          v;
};

__device__ __forceinline__ unsigned int f2bf(float f) {
    unsigned int u = __float_as_uint(f);
    u += 0x7FFFu + ((u >> 16) & 1u);     // round-to-nearest-even
    return u >> 16;
}
__device__ __forceinline__ unsigned int pack2bf(float lo, float hi) {
    return f2bf(lo) | (f2bf(hi) << 16);
}

__device__ __forceinline__ v8f wmma_bf16(const ABFrag& a, const ABFrag& b, v8f c) {
    return __builtin_amdgcn_wmma_f32_16x16x32_bf16(
        /*neg_a=*/false, a.v, /*neg_b=*/false, b.v,
        /*c_mod=*/(short)0, c, /*reuse_a=*/false, /*reuse_b=*/false);
}

__device__ __forceinline__ float gelu_exact(float x) {
    return 0.5f * x * (1.0f + erff(x * 0.70710678118654752f));
}

// ---------------------------------------------------------------------------
// Generic GEMM:  C[M,N] = epi( A[M,K] @ W[N,K]^T + bias )   (fp32 I/O, bf16
// WMMA math, f32 accumulate).  A may be an implicit concat: k < K0 -> A0
// (stride K0), else A1 (stride K-K0).
// EPI: 0 = none, 1 = +bias, 2 = gelu(+bias), 3 = gelu(+bias) + pe[t = row/B]
// Block: 256 thr = 8 waves (4x2), block tile 128x128, wave tile 32x64,
// K step 32.  Software-pipelined: global loads for tile k+1 issue before the
// WMMA burst for tile k.
// ---------------------------------------------------------------------------
template <int EPI>
__global__ __launch_bounds__(256)
void gemm_bf16_wmma(const float* __restrict__ A0, const float* __restrict__ A1,
                    int K0,
                    const float* __restrict__ W, const float* __restrict__ bias,
                    const float* __restrict__ pe,
                    float* __restrict__ C, int M, int N, int K)
{
    __shared__ unsigned short As[128 * 32];   // [row][k] bf16
    __shared__ unsigned short Ws[128 * 32];   // [n][k]  bf16

    const int tid  = threadIdx.x;
    const int lane = tid & 31;
    const int wave = tid >> 5;
    const int wm   = wave >> 1;      // 0..3  (row group of 32)
    const int wn   = wave & 1;       // 0..1  (col group of 64)
    const int half = lane >> 4;      // ISA half-wave split
    const int l16  = lane & 15;
    const int Mb   = blockIdx.y * 128;
    const int Nb   = blockIdx.x * 128;
    const int K1   = K - K0;

    // staging coordinates: thread covers rows (tid>>3)+p*32, cols (tid&7)*4..+3
    const int sr = tid >> 3;
    const int sc = (tid & 7) * 4;

    v8f acc[2][4];
    #pragma unroll
    for (int i = 0; i < 2; ++i)
        #pragma unroll
        for (int j = 0; j < 4; ++j)
            #pragma unroll
            for (int r = 0; r < 8; ++r) acc[i][j][r] = 0.0f;

    float4 ra[4], rw[4];

    auto stage_load = [&](int kb) {
        int gk = kb + sc;
        #pragma unroll
        for (int p = 0; p < 4; ++p) {
            int r = sr + p * 32;
            const float* srcA = (gk < K0)
                ? &A0[(size_t)(Mb + r) * K0 + gk]
                : &A1[(size_t)(Mb + r) * K1 + (gk - K0)];
            ra[p] = *(const float4*)srcA;
            rw[p] = *(const float4*)&W[(size_t)(Nb + r) * K + gk];
        }
    };
    auto stage_store = [&]() {
        #pragma unroll
        for (int p = 0; p < 4; ++p) {
            int r = sr + p * 32;
            uint2 pa, pw;
            pa.x = pack2bf(ra[p].x, ra[p].y);  pa.y = pack2bf(ra[p].z, ra[p].w);
            pw.x = pack2bf(rw[p].x, rw[p].y);  pw.y = pack2bf(rw[p].z, rw[p].w);
            *(uint2*)&As[r * 32 + sc] = pa;
            *(uint2*)&Ws[r * 32 + sc] = pw;
        }
    };

    stage_load(0);
    for (int kb = 0; kb < K; kb += 32) {
        __syncthreads();                      // previous tile's reads done
        stage_store();
        __syncthreads();                      // tile visible to all waves
        if (kb + 32 < K) stage_load(kb + 32); // overlap next global loads
        if (kb + 64 < K)                      // L2 prefetch two tiles ahead
            __builtin_prefetch(&W[(size_t)(Nb + sr) * K + kb + 64], 0, 1);

        // ---- per-wave fragments (ISA 7.12.2 layouts) ----
        ABFrag a[2];
        #pragma unroll
        for (int i = 0; i < 2; ++i) {
            const unsigned short* p = &As[(wm * 32 + i * 16 + l16) * 32];
            a[i].u4[0] = *(const uint4*)(p + half * 8);        // K 0-7 / 8-15
            a[i].u4[1] = *(const uint4*)(p + 16 + half * 8);   // K 16-23 / 24-31
        }
        #pragma unroll
        for (int j = 0; j < 4; ++j) {
            ABFrag b;
            const unsigned short* p = &Ws[(wn * 64 + j * 16 + l16) * 32];
            b.u4[0] = *(const uint4*)(p + half * 8);
            b.u4[1] = *(const uint4*)(p + 16 + half * 8);
            acc[0][j] = wmma_bf16(a[0], b, acc[0][j]);
            acc[1][j] = wmma_bf16(a[1], b, acc[1][j]);
        }
    }

    // ---- epilogue + store (C layout: row = base + 8*half + r, col = l16) ----
    #pragma unroll
    for (int j = 0; j < 4; ++j) {
        int col = Nb + wn * 64 + j * 16 + l16;
        float bv = (EPI >= 1) ? bias[col] : 0.0f;
        #pragma unroll
        for (int i = 0; i < 2; ++i) {
            #pragma unroll
            for (int r = 0; r < 8; ++r) {
                int row = Mb + wm * 32 + i * 16 + half * 8 + r;
                float v = acc[i][j][r] + bv;
                if (EPI >= 2) v = gelu_exact(v);
                if (EPI == 3) v += pe[(size_t)(row / C_B) * N + col];
                C[(size_t)row * N + col] = v;
            }
        }
    }
}

// ---------------------------------------------------------------------------
// Flash attention: grid (T/128, H, B), 256 thr = 8 waves, wave owns 16 rows.
// qkv rows are (t*B+b), cols: q = h*64+dk, k = 1024+h*64+dk, v = 2048+...
//
// Key-axis permutation trick: within a 32-key tile, key k is stored at
// column perm(k) = ((k&15)<<1) | (k>>4) in BOTH the per-wave P tile and the
// transposed V tile.  This lets each lane write its {p0 (key l16), p1 (key
// 16+l16)} pair as ONE packed dword at column 2*l16, while the contraction
// sum_c P[row][c] * Vt[n][c] remains exactly sum_k p[row][k] * V[k][n].
// ---------------------------------------------------------------------------
__global__ __launch_bounds__(256)
void attn_flash(const float* __restrict__ qkv, float* __restrict__ aout)
{
    __shared__ unsigned short Ks[32 * 64];        // [key][dk]
    __shared__ unsigned short Vt[64 * 32];        // [dk][perm(key)]
    __shared__ unsigned short Pw[8 * 16 * 32];    // per-wave P tile [row][perm(key)]

    const int tid  = threadIdx.x;
    const int lane = tid & 31;
    const int wave = tid >> 5;
    const int half = lane >> 4;
    const int l16  = lane & 15;
    const int t0   = blockIdx.x * 128;
    const int h    = blockIdx.y;
    const int b    = blockIdx.z;
    const int qbase = t0 + wave * 16;

    // ---- resident Q fragments (A layout), two K-steps of 32 over DK=64 ----
    ABFrag qf[2];
    {
        const float* qp = qkv + ((size_t)(qbase + l16) * C_B + b) * (3 * C_D) + h * C_DK;
        #pragma unroll
        for (int kk = 0; kk < 2; ++kk) {
            #pragma unroll
            for (int i = 0; i < 8; ++i) {
                qf[kk].h[i]     = (unsigned short)f2bf(qp[kk * 32 + half * 8 + i]);
                qf[kk].h[8 + i] = (unsigned short)f2bf(qp[kk * 32 + 16 + half * 8 + i]);
            }
        }
    }

    v8f o[4];
    #pragma unroll
    for (int s = 0; s < 4; ++s)
        #pragma unroll
        for (int r = 0; r < 8; ++r) o[s][r] = 0.0f;
    float mrow[8], lrow[8];
    #pragma unroll
    for (int r = 0; r < 8; ++r) { mrow[r] = C_NEGINF; lrow[r] = 0.0f; }

    const int kend = t0 + 128;                    // causal upper bound
    for (int kt = 0; kt < kend; kt += 32) {
        __syncthreads();
        // ---- stage K (row major) and V (transposed + key-permuted) ----
        #pragma unroll
        for (int i = 0; i < 8; ++i) {
            int idx = i * 256 + tid;
            int key = idx >> 6, dk = idx & 63;
            int kp  = ((key & 15) << 1) | (key >> 4);      // perm(key)
            size_t base = ((size_t)(kt + key) * C_B + b) * (3 * C_D) + h * C_DK + dk;
            Ks[key * 64 + dk] = (unsigned short)f2bf(qkv[base + C_D]);
            Vt[dk * 32 + kp]  = (unsigned short)f2bf(qkv[base + 2 * C_D]);
        }
        __syncthreads();

        // ---- S = Q K^T : two 16-key subtiles, accumulate both dk halves ----
        v8f s[2];
        #pragma unroll
        for (int nb = 0; nb < 2; ++nb) {
            #pragma unroll
            for (int r = 0; r < 8; ++r) s[nb][r] = 0.0f;
            #pragma unroll
            for (int kk = 0; kk < 2; ++kk) {
                ABFrag bk;
                const unsigned short* p = &Ks[(nb * 16 + l16) * 64 + kk * 32];
                bk.u4[0] = *(const uint4*)(p + half * 8);
                bk.u4[1] = *(const uint4*)(p + 16 + half * 8);
                s[nb] = wmma_bf16(qf[kk], bk, s[nb]);
            }
        }

        // ---- causal mask, scale, online softmax update ----
        #pragma unroll
        for (int r = 0; r < 8; ++r) {
            int qi = qbase + half * 8 + r;
            float s0 = s[0][r] * C_SCALE;                  // key kt + l16
            float s1 = s[1][r] * C_SCALE;                  // key kt + 16 + l16
            if (kt + l16 > qi)      s0 = C_NEGINF;
            if (kt + 16 + l16 > qi) s1 = C_NEGINF;
            float mx = fmaxf(s0, s1);
            #pragma unroll
            for (int d = 8; d >= 1; d >>= 1) mx = fmaxf(mx, __shfl_xor(mx, d, 16));
            float mnew = fmaxf(mrow[r], mx);
            float al   = __expf(mrow[r] - mnew);
            float p0   = __expf(s0 - mnew);
            float p1   = __expf(s1 - mnew);
            float sum  = p0 + p1;
            #pragma unroll
            for (int d = 8; d >= 1; d >>= 1) sum += __shfl_xor(sum, d, 16);
            lrow[r] = lrow[r] * al + sum;
            mrow[r] = mnew;
            // rescale accumulators for this row
            #pragma unroll
            for (int sub = 0; sub < 4; ++sub) o[sub][r] *= al;
            // P tile store: key l16 -> col 2*l16, key 16+l16 -> col 2*l16+1
            // (= perm(key)), one packed dword per row per lane.
            *(unsigned int*)&Pw[wave * 512 + (half * 8 + r) * 32 + l16 * 2] =
                pack2bf(p0, p1);
        }

        // ---- reload P as A-fragment (rows l16, k-dim = permuted key axis) ----
        ABFrag pf;
        {
            const unsigned short* pb = &Pw[wave * 512 + l16 * 32];
            pf.u4[0] = *(const uint4*)(pb + half * 8);
            pf.u4[1] = *(const uint4*)(pb + 16 + half * 8);
        }
        // ---- O += P @ V  (4 dk subtiles of 16; Vt shares the key perm) ----
        #pragma unroll
        for (int sub = 0; sub < 4; ++sub) {
            ABFrag bv;
            const unsigned short* vp = &Vt[(sub * 16 + l16) * 32];
            bv.u4[0] = *(const uint4*)(vp + half * 8);
            bv.u4[1] = *(const uint4*)(vp + 16 + half * 8);
            o[sub] = wmma_bf16(pf, bv, o[sub]);
        }
    }

    // ---- normalize and write: aout[(t*B+b)*D + h*64 + sub*16 + l16] ----
    #pragma unroll
    for (int sub = 0; sub < 4; ++sub) {
        int col = h * C_DK + sub * 16 + l16;
        #pragma unroll
        for (int r = 0; r < 8; ++r) {
            int t = qbase + half * 8 + r;
            aout[((size_t)t * C_B + b) * C_D + col] = o[sub][r] / lrow[r];
        }
    }
}

// ---------------------------------------------------------------------------
// LayerNorm over D=1024, one block (256 thr) per row.
// ---------------------------------------------------------------------------
__global__ __launch_bounds__(256)
void ln_kernel(const float* __restrict__ x, const float* __restrict__ g,
               const float* __restrict__ bta, float* __restrict__ out)
{
    __shared__ float red[2][8];
    const int row = blockIdx.x, tid = threadIdx.x;
    const int lane = tid & 31, wave = tid >> 5;
    const float* xr = x + (size_t)row * C_D;

    float v[4], s = 0.0f, s2 = 0.0f;
    #pragma unroll
    for (int i = 0; i < 4; ++i) {
        v[i] = xr[tid + i * 256];
        s += v[i]; s2 += v[i] * v[i];
    }
    #pragma unroll
    for (int d = 16; d >= 1; d >>= 1) { s += __shfl_xor(s, d); s2 += __shfl_xor(s2, d); }
    if (lane == 0) { red[0][wave] = s; red[1][wave] = s2; }
    __syncthreads();
    float ts = 0.0f, ts2 = 0.0f;
    #pragma unroll
    for (int w = 0; w < 8; ++w) { ts += red[0][w]; ts2 += red[1][w]; }
    float mu  = ts * (1.0f / C_D);
    float var = ts2 * (1.0f / C_D) - mu * mu;
    float inv = rsqrtf(var + 1e-6f);
    #pragma unroll
    for (int i = 0; i < 4; ++i) {
        int c = tid + i * 256;
        out[(size_t)row * C_D + c] = (v[i] - mu) * inv * g[c] + bta[c];
    }
}

// ---------------------------------------------------------------------------
// GRU gate combine: x = (1-z)*x + z*tanh(gh*r), r=sig(gr), z=sig(gz-bg).
// gbuf rows stride 3*D.  Grid = M*D/256.
// ---------------------------------------------------------------------------
__global__ __launch_bounds__(256)
void gru_combine(float* __restrict__ x, const float* __restrict__ gbuf,
                 const float* __restrict__ bg)
{
    int i = blockIdx.x * 256 + threadIdx.x;
    int m = i >> 10, c = i & 1023;
    const float* gm = gbuf + (size_t)m * (3 * C_D);
    float gr = gm[c], gz = gm[C_D + c], gh = gm[2 * C_D + c];
    float r  = 1.0f / (1.0f + __expf(-gr));
    float z  = 1.0f / (1.0f + __expf(-(gz - bg[c])));
    float hc = tanhf(gh * r);
    x[i] = (1.0f - z) * x[i] + z * hc;
}

// ---------------------------------------------------------------------------
// Host orchestration.  d_ws layout (floats):
//   x: [0, 2M)   h: [2M, 4M)   g/qkv: [4M, 10M)   ff: [10M, 18M)  => 72 MB
// ---------------------------------------------------------------------------
extern "C" void kernel_launch(void* const* d_in, const int* in_sizes, int n_in,
                              void* d_out, int out_size, void* d_ws, size_t ws_size,
                              hipStream_t stream)
{
    const float* xin   = (const float*)d_in[0];
    const float* pe    = (const float*)d_in[1];
    const float* in_w  = (const float*)d_in[2];
    const float* in_b  = (const float*)d_in[3];
    const float* qkv_w = (const float*)d_in[4];
    const float* out_w = (const float*)d_in[5];
    const float* out_b = (const float*)d_in[6];
    const float* ln1_g = (const float*)d_in[7];
    const float* ln1_b = (const float*)d_in[8];
    const float* ln2_g = (const float*)d_in[9];
    const float* ln2_b = (const float*)d_in[10];
    const float* ff_w1 = (const float*)d_in[11];
    const float* ff_b1 = (const float*)d_in[12];
    const float* ff_w2 = (const float*)d_in[13];
    const float* ff_b2 = (const float*)d_in[14];
    const float* g1_w  = (const float*)d_in[15];
    const float* g1_bg = (const float*)d_in[16];
    const float* g2_w  = (const float*)d_in[17];
    const float* g2_bg = (const float*)d_in[18];
    const float* on_g  = (const float*)d_in[19];
    const float* on_b  = (const float*)d_in[20];

    float* xbuf = (float*)d_ws;
    float* hbuf = xbuf + (size_t)C_M * C_D;           // 2M floats
    float* gbuf = hbuf + (size_t)C_M * C_D;           // 6M floats (qkv / gates)
    float* fbuf = gbuf + (size_t)C_M * 3 * C_D;       // 8M floats (ff / attn-proj)

    const dim3 blk(256);
    const dim3 gLN(C_M);
    const dim3 gEL((C_M * C_D) / 256);
    const dim3 gAT(C_T / 128, C_H, C_B);
    auto gG = [](int N) { return dim3(N / 128, C_M / 128); };

    // input projection: x = gelu(xin @ in_w^T + in_b) + pe
    gemm_bf16_wmma<3><<<gG(C_D), blk, 0, stream>>>(
        xin, xin, C_D, in_w, in_b, pe, xbuf, C_M, C_D, C_D);

    for (int l = 0; l < C_L; ++l) {
        const float* qw  = qkv_w + (size_t)l * 3 * C_D * C_D;
        const float* ow  = out_w + (size_t)l * C_D * C_D;
        const float* f1  = ff_w1 + (size_t)l * C_FF * C_D;
        const float* f2  = ff_w2 + (size_t)l * C_D * C_FF;
        const float* gw1 = g1_w  + (size_t)l * 3 * C_D * 2 * C_D;
        const float* gw2 = g2_w  + (size_t)l * 3 * C_D * 2 * C_D;

        // h = LN1(x)
        ln_kernel<<<gLN, blk, 0, stream>>>(xbuf, ln1_g + l * C_D, ln1_b + l * C_D, hbuf);
        // qkv = h @ qw^T
        gemm_bf16_wmma<0><<<gG(3 * C_D), blk, 0, stream>>>(
            hbuf, hbuf, C_D, qw, nullptr, nullptr, gbuf, C_M, 3 * C_D, C_D);
        // h = attention(qkv)
        attn_flash<<<gAT, blk, 0, stream>>>(gbuf, hbuf);
        // f = h @ ow^T + ob
        gemm_bf16_wmma<1><<<gG(C_D), blk, 0, stream>>>(
            hbuf, hbuf, C_D, ow, out_b + l * C_D, nullptr, fbuf, C_M, C_D, C_D);
        // g = concat(x, f) @ gw1^T ; x = gru(x, g)
        gemm_bf16_wmma<0><<<gG(3 * C_D), blk, 0, stream>>>(
            xbuf, fbuf, C_D, gw1, nullptr, nullptr, gbuf, C_M, 3 * C_D, 2 * C_D);
        gru_combine<<<gEL, blk, 0, stream>>>(xbuf, gbuf, g1_bg + l * C_D);
        // h = LN2(x)
        ln_kernel<<<gLN, blk, 0, stream>>>(xbuf, ln2_g + l * C_D, ln2_b + l * C_D, hbuf);
        // f = gelu(h @ f1^T + b1) ; h = f @ f2^T + b2
        gemm_bf16_wmma<2><<<gG(C_FF), blk, 0, stream>>>(
            hbuf, hbuf, C_D, f1, ff_b1 + l * C_FF, nullptr, fbuf, C_M, C_FF, C_D);
        gemm_bf16_wmma<1><<<gG(C_D), blk, 0, stream>>>(
            fbuf, fbuf, C_FF, f2, ff_b2 + l * C_D, nullptr, hbuf, C_M, C_D, C_FF);
        // g = concat(x, h) @ gw2^T ; x = gru(x, g)
        gemm_bf16_wmma<0><<<gG(3 * C_D), blk, 0, stream>>>(
            xbuf, hbuf, C_D, gw2, nullptr, nullptr, gbuf, C_M, 3 * C_D, 2 * C_D);
        gru_combine<<<gEL, blk, 0, stream>>>(xbuf, gbuf, g2_bg + l * C_D);
    }

    // final layer norm -> output
    ln_kernel<<<gLN, blk, 0, stream>>>(xbuf, on_g, on_b, (float*)d_out);
}
